// EdgeAttention_53944789238364
// MI455X (gfx1250) — matrix operand order repaired and checked by reading, not compiled, not run on hardware
//
#include <hip/hip_runtime.h>

// ---------------- types for CDNA5 WMMA ----------------
typedef __attribute__((ext_vector_type(16))) __bf16 v16bf;
typedef __attribute__((ext_vector_type(8)))  __bf16 v8bf;
typedef __attribute__((ext_vector_type(4)))  __bf16 v4bf;
typedef __attribute__((ext_vector_type(8)))  float  v8f;

constexpr int E_EDGES = 50000;
constexpr int DIM     = 128;
constexpr int NHEADS  = 8;
constexpr int HDIM    = 16;
constexpr float NORM  = 0.25f;   // HEAD_DIM^-0.5 = 16^-0.5

// ---------------- helpers ----------------
__device__ __forceinline__ void atomicMaxF32(float* addr, float val) {
  // classic sign-split trick: works with -inf init, native int atomics
  if (val >= 0.0f) {
    atomicMax((int*)addr, __float_as_int(val));
  } else {
    atomicMin((unsigned int*)addr, __float_as_uint(val));
  }
}

// ---------------- f32 -> bf16 bulk convert (vectorized x4) ----------------
__global__ void cvt_f32_to_bf16(const float* __restrict__ src,
                                __bf16* __restrict__ dst, long n4) {
  long i = (long)blockIdx.x * blockDim.x + threadIdx.x;
  if (i >= n4) return;
  float4 f = ((const float4*)src)[i];
  v4bf o;
  o[0] = (__bf16)f.x; o[1] = (__bf16)f.y; o[2] = (__bf16)f.z; o[3] = (__bf16)f.w;
  ((v4bf*)dst)[i] = o;
}

// ---------------- 128x128 weight: transpose + convert to bf16 (N-major) ---
__global__ void cvt_w_transpose(const float* __restrict__ W,
                                __bf16* __restrict__ WT) {
  int n = blockIdx.x;     // output row  (column of W)
  int k = threadIdx.x;    // output col  (row of W)
  WT[n * DIM + k] = (__bf16)W[k * DIM + n];
}

// ---------------- GEMM: C[M x 128] = A[M x 128] * WT^T, bf16 WMMA --------
// A row-major bf16, WT is N-major bf16 (WT[n][k] = W[k][n]).
// Block = 256 threads = 8 waves; wave w owns column tile w (8 * 16 = 128).
// blockIdx.x = row tile (M/16). K loop: 4 x wmma_f32_16x16x32_bf16.
__global__ void gemm_bf16_wmma(const __bf16* __restrict__ A,
                               const __bf16* __restrict__ WT,
                               float* __restrict__ C,
                               const float* __restrict__ bias,
                               float scale) {
  const int lane  = threadIdx.x & 31;
  const int wave  = threadIdx.x >> 5;
  const int lmod  = lane & 15;
  const int lhalf = lane >> 4;
  const int row0  = blockIdx.x << 4;
  const int col0  = wave << 4;

  const __bf16* arow = A  + (size_t)(row0 + lmod) * DIM + lhalf * 8;
  const __bf16* brow = WT + (size_t)(col0 + lmod) * DIM + lhalf * 16;

  v8f c = {};
#pragma unroll
  for (int kb = 0; kb < 4; ++kb) {
    // A fragment: lanes 0-15 rows, K = kb*32 + lhalf*16 + 0..15
    v8bf alo = *(const v8bf*)(arow + kb * 32 + lhalf * 8);
    v8bf ahi = *(const v8bf*)(arow + kb * 32 + lhalf * 8 + 8);
    v16bf a = __builtin_shufflevector(alo, ahi,
        0,1,2,3,4,5,6,7,8,9,10,11,12,13,14,15);
    // B fragment: lane holds column col0+lmod, 16 contiguous K values
    v16bf b = *(const v16bf*)(brow + kb * 32);
    c = __builtin_amdgcn_wmma_f32_16x16x32_bf16(
            false, a, false, b, (short)0, c, false, false);
  }

  const float badd = bias ? bias[col0 + lmod] : 0.0f;
  float* crow = C + (size_t)(row0 + lhalf * 8) * DIM + col0 + lmod;
#pragma unroll
  for (int v = 0; v < 8; ++v) {
    crow[(size_t)v * DIM] = c[v] * scale + badd;
  }
}

// ---------------- init m/denom/o --------------------------------------
__global__ void init_buffers(float* __restrict__ o, float* __restrict__ m,
                             float* __restrict__ denom, long n_o, long n_mh) {
  long i = (long)blockIdx.x * blockDim.x + threadIdx.x;
  if (i < n_o)  o[i] = 0.0f;
  if (i < n_mh) { m[i] = -__builtin_inff(); denom[i] = 0.0f; }
}

// ---------------- pair logits + segment max ---------------------------
__global__ void attn_logits(const float* __restrict__ qh,
                            const float* __restrict__ kh,
                            const int* __restrict__ ee,
                            float* __restrict__ attn,
                            float* __restrict__ m, int total) {
  int t = blockIdx.x * blockDim.x + threadIdx.x;
  if (t >= total) return;
  int p = t >> 3, h = t & 7;
  int sg = ee[2 * p], nb = ee[2 * p + 1];
  const float4* qp = (const float4*)(qh + (size_t)sg * DIM + h * HDIM);
  const float4* kp = (const float4*)(kh + (size_t)nb * DIM + h * HDIM);
  float s = 0.0f;
#pragma unroll
  for (int i = 0; i < 4; ++i) {
    float4 a = qp[i], b = kp[i];
    s += a.x * b.x + a.y * b.y + a.z * b.z + a.w * b.w;
  }
  attn[t] = s;
  atomicMaxF32(&m[(size_t)sg * NHEADS + h], s);
}

// ---------------- exp(attn - max) + segment sum -----------------------
__global__ void attn_exp(const int* __restrict__ ee,
                         float* __restrict__ attn,
                         const float* __restrict__ m,
                         float* __restrict__ denom, int total) {
  int t = blockIdx.x * blockDim.x + threadIdx.x;
  if (t >= total) return;
  int p = t >> 3, h = t & 7;
  int sg = ee[2 * p];
  float w = __expf(attn[t] - m[(size_t)sg * NHEADS + h]);
  attn[t] = w;
  unsafeAtomicAdd(&denom[(size_t)sg * NHEADS + h], w);
}

// ---------------- normalize + weighted scatter of vh ------------------
__global__ void attn_scatter(const int* __restrict__ ee,
                             const float* __restrict__ attn,
                             const float* __restrict__ denom,
                             const float* __restrict__ vh,
                             float* __restrict__ o, int total) {
  int t = blockIdx.x * blockDim.x + threadIdx.x;
  if (t >= total) return;
  int p = t >> 3, h = t & 7;
  int sg = ee[2 * p], nb = ee[2 * p + 1];
  float w = attn[t] / denom[(size_t)sg * NHEADS + h];
  const float4* vp = (const float4*)(vh + (size_t)nb * DIM + h * HDIM);
  float* op = o + (size_t)sg * DIM + h * HDIM;
#pragma unroll
  for (int i = 0; i < 4; ++i) {
    float4 vv = vp[i];
    unsafeAtomicAdd(op + 4 * i + 0, w * vv.x);
    unsafeAtomicAdd(op + 4 * i + 1, w * vv.y);
    unsafeAtomicAdd(op + 4 * i + 2, w * vv.z);
    unsafeAtomicAdd(op + 4 * i + 3, w * vv.w);
  }
}

// ======================================================================
extern "C" void kernel_launch(void* const* d_in, const int* in_sizes, int n_in,
                              void* d_out, int out_size, void* d_ws, size_t ws_size,
                              hipStream_t stream) {
  const float* q  = (const float*)d_in[0];
  const float* k  = (const float*)d_in[1];
  const float* v  = (const float*)d_in[2];
  const int*   ee = (const int*)d_in[3];
  const float* Wq = (const float*)d_in[4];
  const float* Wk = (const float*)d_in[5];
  const float* Wv = (const float*)d_in[6];
  const float* Wo = (const float*)d_in[7];
  const float* bo = (const float*)d_in[8];
  float* out = (float*)d_out;

  const int P = in_sizes[3] / 2;           // number of (seg, nbr) pairs
  const long ED   = (long)E_EDGES * DIM;   // 6.4M
  const long EH   = (long)E_EDGES * NHEADS;

  // ---- workspace layout (256B aligned) ----
  char* ws = (char*)d_ws;
  auto alloc = [&](size_t bytes) -> char* {
    char* p = ws; ws += (bytes + 255) & ~(size_t)255; return p;
  };
  __bf16* qb  = (__bf16*)alloc(ED * 2);         // bf16 q   (reused later for o_bf16)
  __bf16* kb  = (__bf16*)alloc(ED * 2);
  __bf16* vb  = (__bf16*)alloc(ED * 2);
  __bf16* WqT = (__bf16*)alloc(DIM * DIM * 2);
  __bf16* WkT = (__bf16*)alloc(DIM * DIM * 2);
  __bf16* WvT = (__bf16*)alloc(DIM * DIM * 2);
  __bf16* WoT = (__bf16*)alloc(DIM * DIM * 2);
  float*  qh  = (float*)alloc(ED * 4);
  float*  kh  = (float*)alloc(ED * 4);
  float*  vh  = (float*)alloc(ED * 4);
  float*  oacc= (float*)alloc(ED * 4);
  float*  m   = (float*)alloc(EH * 4);
  float*  den = (float*)alloc(EH * 4);
  float*  att = (float*)alloc((size_t)P * NHEADS * 4);

  const int TB = 256;
  const long n4 = ED / 4;
  const int cvt_blocks = (int)((n4 + TB - 1) / TB);

  // 1) convert activations to bf16
  cvt_f32_to_bf16<<<cvt_blocks, TB, 0, stream>>>(q, qb, n4);
  cvt_f32_to_bf16<<<cvt_blocks, TB, 0, stream>>>(k, kb, n4);
  cvt_f32_to_bf16<<<cvt_blocks, TB, 0, stream>>>(v, vb, n4);

  // 2) transpose+convert weights (N-major bf16 for WMMA B-operand loads)
  cvt_w_transpose<<<DIM, DIM, 0, stream>>>(Wq, WqT);
  cvt_w_transpose<<<DIM, DIM, 0, stream>>>(Wk, WkT);
  cvt_w_transpose<<<DIM, DIM, 0, stream>>>(Wv, WvT);
  cvt_w_transpose<<<DIM, DIM, 0, stream>>>(Wo, WoT);

  // 3) projection GEMMs (NORM folded into qh scale)
  const int row_tiles = E_EDGES / 16;      // 3125, exact
  gemm_bf16_wmma<<<row_tiles, TB, 0, stream>>>(qb, WqT, qh, nullptr, NORM);
  gemm_bf16_wmma<<<row_tiles, TB, 0, stream>>>(kb, WkT, kh, nullptr, 1.0f);
  gemm_bf16_wmma<<<row_tiles, TB, 0, stream>>>(vb, WvT, vh, nullptr, 1.0f);

  // 4) init accumulators (m=-inf, denom=0, o=0)
  init_buffers<<<(int)((ED + TB - 1) / TB), TB, 0, stream>>>(oacc, m, den, ED, EH);

  // 5) softmax-attention over pairs
  const int total = P * NHEADS;
  const int pb = (total + TB - 1) / TB;
  attn_logits <<<pb, TB, 0, stream>>>(qh, kh, ee, att, m, total);
  attn_exp    <<<pb, TB, 0, stream>>>(ee, att, m, den, total);
  attn_scatter<<<pb, TB, 0, stream>>>(ee, att, den, vh, oacc, total);

  // 6) final GEMM: out = o @ Wo + bo   (reuse qb as bf16 staging for o)
  cvt_f32_to_bf16<<<cvt_blocks, TB, 0, stream>>>(oacc, qb, n4);
  gemm_bf16_wmma<<<row_tiles, TB, 0, stream>>>(qb, WoT, out, bo, 1.0f);
}
